// GraphSAGEEdgeClassifier_20633022890439
// MI455X (gfx1250) — matrix operand order repaired and checked by reading, not compiled, not run on hardware
//
#include <hip/hip_runtime.h>

#define N_NODES 100000
#define N_EDGES 1600000
#define D_IN 128
#define HID 128
#define E_CH 16

typedef __attribute__((ext_vector_type(16))) __bf16        v16bf;
typedef __attribute__((ext_vector_type(8)))  float         v8f;
typedef __attribute__((ext_vector_type(4)))  unsigned int  u32x4;
typedef __attribute__((ext_vector_type(8)))  unsigned int  u32x8;

// ---------------------------------------------------------------- helpers

__device__ __forceinline__ unsigned short f2bf(float f) {
  unsigned int u = __builtin_bit_cast(unsigned int, f);
  u += 0x7FFFu + ((u >> 16) & 1u);           // round-to-nearest-even
  return (unsigned short)(u >> 16);
}

// A fragment (16x32 bf16, MxK) from an LDS tile with row stride `stride`
// (in bf16 elements), rows [rowbase, rowbase+16). ISA layout: lane L ->
// row M = L&15; elements 0..7 at K = kbase + (L>>4)*8 + e, elems 8..15 at +16.
__device__ __forceinline__ v16bf load_a_frag(const unsigned short* lds,
                                             int stride, int rowbase,
                                             int kbase, int lane) {
  int m = (lane & 15) + rowbase, half = lane >> 4;
  const unsigned short* p = lds + m * stride + kbase + half * 8;
  u32x4 lo = *(const u32x4*)p;          // K = kbase + half*8 .. +7
  u32x4 hi = *(const u32x4*)(p + 16);   // K = kbase + 16 + half*8 .. +7
  u32x8 u;
  u[0]=lo[0]; u[1]=lo[1]; u[2]=lo[2]; u[3]=lo[3];
  u[4]=hi[0]; u[5]=hi[1]; u[6]=hi[2]; u[7]=hi[3];
  return __builtin_bit_cast(v16bf, u);
}

// B fragment (32x16 bf16, KxN) from a fragment-major packed weight buffer:
// each fragment = 32 lanes x 16 contiguous bf16 (32B per lane).
__device__ __forceinline__ v16bf load_b_frag(const unsigned short* pack,
                                             int frag, int lane) {
  const u32x8* p = (const u32x8*)pack + (size_t)frag * 32 + lane;
  return __builtin_bit_cast(v16bf, *p);
}

// ---------------------------------------------------------------- kernels

__global__ void zero_kernel(float* __restrict__ p, int n) {
  int i = (blockIdx.x * blockDim.x + threadIdx.x) * 4;
  if (i < n) { p[i] = 0.f; p[i+1] = 0.f; p[i+2] = 0.f; p[i+3] = 0.f; }
}

// Pack B = [A ; Bsrc] (rows: [0,split) from A, [split,Kreal) from Bsrc,
// [Kreal,KT*32) zero) of shape (KT*32 x N) into bf16 fragment-major layout.
__global__ void pack_wb_kernel(const float* __restrict__ A,
                               const float* __restrict__ Bsrc,
                               int split, int Kreal, int N, int NT,
                               unsigned short* __restrict__ out, int total) {
  int t = blockIdx.x * blockDim.x + threadIdx.x;
  if (t >= total) return;
  int f    = t >> 9;          // 512 elements per fragment
  int w    = t & 511;
  int lane = w >> 4;
  int i    = w & 15;
  int kt = f / NT, nt = f % NT;
  int k = kt * 32 + (lane >> 4) * 8 + (i & 7) + (i >> 3) * 16;
  int n = nt * 16 + (lane & 15);
  float v = 0.f;
  if (k < split)      v = A[(size_t)k * N + n];
  else if (k < Kreal) v = Bsrc[(size_t)(k - split) * N + n];
  out[t] = f2bf(v);
}

__global__ void deg_kernel(const int* __restrict__ ei, float* __restrict__ deg) {
  int e = blockIdx.x * blockDim.x + threadIdx.x;
  if (e < N_EDGES) atomicAdd(deg + ei[N_EDGES + e], 1.0f);
}

// scatter-sum: msg[dst] += feat[src]; one thread per (edge, 4 floats)
__global__ void scatter_kernel(const float* __restrict__ feat,
                               const int* __restrict__ ei,
                               float* __restrict__ msg) {
  int gid = blockIdx.x * blockDim.x + threadIdx.x;
  int e = gid >> 5;
  if (e >= N_EDGES) return;
  int c = (gid & 31) * 4;
  int s = ei[e];
  int d = ei[N_EDGES + e];
  float4 v = *(const float4*)(feat + (size_t)s * 128 + c);
  float* o = msg + (size_t)d * 128 + c;
  atomicAdd(o + 0, v.x); atomicAdd(o + 1, v.y);
  atomicAdd(o + 2, v.z); atomicAdd(o + 3, v.w);
}

// SAGE layer: h = relu( [msg/deg | xin] (32x256) @ Wpack (256x128) + b )
// 256 threads = 8 waves; wave w owns output col tile w; 32 nodes per block;
// each B fragment is reused for two row tiles (2 WMMAs per B fetch).
__global__ __launch_bounds__(256)
void sage_kernel(const float* __restrict__ msg, const float* __restrict__ deg,
                 const float* __restrict__ xin,
                 const unsigned short* __restrict__ wpack,
                 const float* __restrict__ bias, float* __restrict__ hout) {
  __shared__ __align__(16) unsigned short lds_a[32 * 256];
  int tid = threadIdx.x;
  int n0  = blockIdx.x * 32;

  for (int i = tid * 4; i < 32 * 256; i += 256 * 4) {
    int row = i >> 8, c = i & 255;
    int node = n0 + row;
    float4 v;
    if (c < 128) {
      float dg = deg[node];
      float r  = 1.f / (dg > 1.f ? dg : 1.f);
      float4 m4 = *(const float4*)(msg + (size_t)node * 128 + c);
      v.x = m4.x * r; v.y = m4.y * r; v.z = m4.z * r; v.w = m4.w * r;
    } else {
      v = *(const float4*)(xin + (size_t)node * 128 + (c - 128));
    }
    unsigned short* o = lds_a + i;
    o[0] = f2bf(v.x); o[1] = f2bf(v.y); o[2] = f2bf(v.z); o[3] = f2bf(v.w);
  }
  __syncthreads();

  int wave = tid >> 5, lane = tid & 31, half = lane >> 4, nloc = lane & 15;
  v8f acc0 = {}, acc1 = {};
  for (int kt = 0; kt < 8; ++kt) {
    v16bf b  = load_b_frag(wpack, kt * 8 + wave, lane);
    v16bf a0 = load_a_frag(lds_a, 256, 0,  kt * 32, lane);
    v16bf a1 = load_a_frag(lds_a, 256, 16, kt * 32, lane);
    acc0 = __builtin_amdgcn_wmma_f32_16x16x32_bf16(false, a0, false, b,
                                                   (short)0, acc0, false, false);
    acc1 = __builtin_amdgcn_wmma_f32_16x16x32_bf16(false, a1, false, b,
                                                   (short)0, acc1, false, false);
  }
  int col = wave * 16 + nloc;
  float bf = bias[col];
#pragma unroll
  for (int r = 0; r < 8; ++r) {
    int m = half * 8 + r;
    float v0 = acc0[r] + bf;
    float v1 = acc1[r] + bf;
    hout[(size_t)(n0 + m) * 128 + col]      = v0 > 0.f ? v0 : 0.f;
    hout[(size_t)(n0 + 16 + m) * 128 + col] = v1 > 0.f ? v1 : 0.f;
  }
}

// Fused edge classifier: 32 edges per block.
// feat = [h[src] | h[dst] | eattr | 0pad]  (32 x 288 bf16)
// z1 = relu(feat @ Wc1p + bc1)  (32x128, 8 waves, 2 row tiles per wave)
// z2 = relu(z1 @ Wc2p + bc2)    (32x64, 4 waves, 2 row tiles per wave)
// out = sigmoid(z2 @ wc3 + bc3)
__global__ __launch_bounds__(256)
void edge_cls_kernel(const float* __restrict__ h, const int* __restrict__ ei,
                     const float* __restrict__ eattr,
                     const unsigned short* __restrict__ wc1p,
                     const unsigned short* __restrict__ wc2p,
                     const float* __restrict__ wc3,
                     const float* __restrict__ bc1, const float* __restrict__ bc2,
                     const float* __restrict__ bc3, float* __restrict__ out) {
  __shared__ __align__(16) unsigned short lds_a[32 * 288];
  __shared__ __align__(16) unsigned short lds_z1[32 * 128];
  __shared__ __align__(16) float lds_z2[32 * 64];
  __shared__ int lds_sd[64];

  int tid = threadIdx.x;
  int e0  = blockIdx.x * 32;
  if (tid < 32) {
    lds_sd[tid]      = ei[e0 + tid];
    lds_sd[32 + tid] = ei[N_EDGES + e0 + tid];
  }
  __syncthreads();

  for (int i = tid * 4; i < 32 * 288; i += 256 * 4) {
    int row = i / 288, c = i % 288;
    float4 v; v.x = 0.f; v.y = 0.f; v.z = 0.f; v.w = 0.f;
    if (c < 128)
      v = *(const float4*)(h + (size_t)lds_sd[row] * 128 + c);
    else if (c < 256)
      v = *(const float4*)(h + (size_t)lds_sd[32 + row] * 128 + (c - 128));
    else if (c < 272)
      v = *(const float4*)(eattr + (size_t)(e0 + row) * 16 + (c - 256));
    unsigned short* o = lds_a + i;
    o[0] = f2bf(v.x); o[1] = f2bf(v.y); o[2] = f2bf(v.z); o[3] = f2bf(v.w);
  }
  __syncthreads();

  int wave = tid >> 5, lane = tid & 31, half = lane >> 4, nloc = lane & 15;

  { // GEMM1: K = 288 (9 k-tiles), all 8 waves, 2 row tiles each
    v8f acc0 = {}, acc1 = {};
    for (int kt = 0; kt < 9; ++kt) {
      v16bf b  = load_b_frag(wc1p, kt * 8 + wave, lane);
      v16bf a0 = load_a_frag(lds_a, 288, 0,  kt * 32, lane);
      v16bf a1 = load_a_frag(lds_a, 288, 16, kt * 32, lane);
      acc0 = __builtin_amdgcn_wmma_f32_16x16x32_bf16(false, a0, false, b,
                                                     (short)0, acc0, false, false);
      acc1 = __builtin_amdgcn_wmma_f32_16x16x32_bf16(false, a1, false, b,
                                                     (short)0, acc1, false, false);
    }
    int col = wave * 16 + nloc;
    float bf = bc1[col];
#pragma unroll
    for (int r = 0; r < 8; ++r) {
      int m = half * 8 + r;
      float v0 = acc0[r] + bf;
      float v1 = acc1[r] + bf;
      lds_z1[m * 128 + col]        = f2bf(v0 > 0.f ? v0 : 0.f);
      lds_z1[(m + 16) * 128 + col] = f2bf(v1 > 0.f ? v1 : 0.f);
    }
  }
  __syncthreads();

  if (wave < 4) { // GEMM2: K = 128 (4 k-tiles), 4 waves, 2 row tiles each
    v8f acc0 = {}, acc1 = {};
    for (int kt = 0; kt < 4; ++kt) {
      v16bf b  = load_b_frag(wc2p, kt * 4 + wave, lane);
      v16bf a0 = load_a_frag(lds_z1, 128, 0,  kt * 32, lane);
      v16bf a1 = load_a_frag(lds_z1, 128, 16, kt * 32, lane);
      acc0 = __builtin_amdgcn_wmma_f32_16x16x32_bf16(false, a0, false, b,
                                                     (short)0, acc0, false, false);
      acc1 = __builtin_amdgcn_wmma_f32_16x16x32_bf16(false, a1, false, b,
                                                     (short)0, acc1, false, false);
    }
    int col = wave * 16 + nloc;
    float bf = bc2[col];
#pragma unroll
    for (int r = 0; r < 8; ++r) {
      int m = half * 8 + r;
      float v0 = acc0[r] + bf;
      float v1 = acc1[r] + bf;
      lds_z2[m * 64 + col]        = v0 > 0.f ? v0 : 0.f;
      lds_z2[(m + 16) * 64 + col] = v1 > 0.f ? v1 : 0.f;
    }
  }
  __syncthreads();

  if (tid < 32) { // final dot + sigmoid
    float s = bc3[0];
#pragma unroll 4
    for (int j = 0; j < 64; ++j) s += lds_z2[tid * 64 + j] * wc3[j];
    out[e0 + tid] = 1.f / (1.f + __expf(-s));
  }
}

// ---------------------------------------------------------------- launch

static inline int cdiv(int a, int b) { return (a + b - 1) / b; }

extern "C" void kernel_launch(void* const* d_in, const int* in_sizes, int n_in,
                              void* d_out, int out_size, void* d_ws, size_t ws_size,
                              hipStream_t stream) {
  const float* x   = (const float*)d_in[0];
  const int*   ei  = (const int*)  d_in[1];
  const float* ea  = (const float*)d_in[2];
  const float* Wl0 = (const float*)d_in[3];
  const float* Wr0 = (const float*)d_in[4];
  const float* b0  = (const float*)d_in[5];
  const float* Wl1 = (const float*)d_in[6];
  const float* Wr1 = (const float*)d_in[7];
  const float* b1  = (const float*)d_in[8];
  const float* Wc1 = (const float*)d_in[9];
  const float* bc1 = (const float*)d_in[10];
  const float* Wc2 = (const float*)d_in[11];
  const float* bc2 = (const float*)d_in[12];
  const float* Wc3 = (const float*)d_in[13];
  const float* bc3 = (const float*)d_in[14];
  float* out = (float*)d_out;

  // workspace carve-out (256B aligned regions)
  char* ws = (char*)d_ws;
  size_t off = 0;
  auto carve = [&](size_t bytes) -> char* {
    off = (off + 255) & ~(size_t)255;
    char* p = ws + off;
    off += bytes;
    return p;
  };
  float* deg = (float*)carve((size_t)N_NODES * 4);
  float* msg = (float*)carve((size_t)N_NODES * 128 * 4);
  float* h0  = (float*)carve((size_t)N_NODES * 128 * 4);
  float* h1  = (float*)carve((size_t)N_NODES * 128 * 4);
  unsigned short* w0p  = (unsigned short*)carve(32768 * 2);
  unsigned short* w1p  = (unsigned short*)carve(32768 * 2);
  unsigned short* wc1p = (unsigned short*)carve(36864 * 2);
  unsigned short* wc2p = (unsigned short*)carve(8192 * 2);

  // pack weights to bf16 fragment-major
  pack_wb_kernel<<<cdiv(32768, 256), 256, 0, stream>>>(Wl0, Wr0, 128, 256, 128, 8, w0p, 32768);
  pack_wb_kernel<<<cdiv(32768, 256), 256, 0, stream>>>(Wl1, Wr1, 128, 256, 128, 8, w1p, 32768);
  pack_wb_kernel<<<cdiv(36864, 256), 256, 0, stream>>>(Wc1, Wc1, 272, 272, 128, 8, wc1p, 36864);
  pack_wb_kernel<<<cdiv(8192, 256), 256, 0, stream>>>(Wc2, Wc2, 128, 128, 64, 4, wc2p, 8192);

  // layer 0
  zero_kernel<<<cdiv(N_NODES / 4, 256), 256, 0, stream>>>(deg, N_NODES);
  zero_kernel<<<cdiv(N_NODES * 128 / 4, 256), 256, 0, stream>>>(msg, N_NODES * 128);
  deg_kernel<<<cdiv(N_EDGES, 256), 256, 0, stream>>>(ei, deg);
  scatter_kernel<<<cdiv(N_EDGES * 32, 256), 256, 0, stream>>>(x, ei, msg);
  sage_kernel<<<N_NODES / 32, 256, 0, stream>>>(msg, deg, x, w0p, b0, h0);

  // layer 1
  zero_kernel<<<cdiv(N_NODES * 128 / 4, 256), 256, 0, stream>>>(msg, N_NODES * 128);
  scatter_kernel<<<cdiv(N_EDGES * 32, 256), 256, 0, stream>>>(h0, ei, msg);
  sage_kernel<<<N_NODES / 32, 256, 0, stream>>>(msg, deg, h0, w1p, b1, h1);

  // fused edge classifier
  edge_cls_kernel<<<N_EDGES / 32, 256, 0, stream>>>(h1, ei, ea, wc1p, wc2p, Wc3,
                                                    bc1, bc2, bc3, out);
}
